// RNNModel_39075612459431
// MI455X (gfx1250) — compile-verified, matrix operand order checked by reading, not run on hardware
//
#include <hip/hip_runtime.h>

// ---------------------------------------------------------------------------
// Problem constants
// ---------------------------------------------------------------------------
constexpr int SEQ   = 512;
constexpr int BATCH = 64;
constexpr int EMB   = 1024;
constexpr int HID   = 1024;
constexpr int H3    = 3 * HID;   // 3072
constexpr int NSLOT = 3;

// ---------------------------------------------------------------------------
// CDNA5 async global->LDS staging (guarded; falls back to sync copies)
// ---------------------------------------------------------------------------
#if __has_builtin(__builtin_amdgcn_global_load_async_to_lds_b128) && \
    __has_builtin(__builtin_amdgcn_s_wait_asynccnt)
#define HAVE_ASYNC_LDS 1
#else
#define HAVE_ASYNC_LDS 0
#endif

// ---------------------------------------------------------------------------
// Types / small helpers
// ---------------------------------------------------------------------------
typedef __bf16 bf16x16 __attribute__((ext_vector_type(16)));
typedef float  floatx8 __attribute__((ext_vector_type(8)));
typedef int    intx4   __attribute__((ext_vector_type(4)));
typedef __attribute__((address_space(1))) intx4 gintx4;   // global v4i32
typedef __attribute__((address_space(3))) intx4 lintx4;   // LDS    v4i32

__device__ __forceinline__ floatx8 fzero8() {
    floatx8 z = {0.f, 0.f, 0.f, 0.f, 0.f, 0.f, 0.f, 0.f};
    return z;
}

__device__ __forceinline__ unsigned short f2bf(float x) {
    unsigned u = __float_as_uint(x);
    u += 0x7FFFu + ((u >> 16) & 1u);        // round-to-nearest-even
    return (unsigned short)(u >> 16);
}
__device__ __forceinline__ float bf2f(unsigned short b) {
    return __uint_as_float(((unsigned)b) << 16);
}
__device__ __forceinline__ float sigm(float x) {
    return 1.f / (1.f + __expf(-x));
}

// Copy 16 bytes global -> LDS.  Async (ASYNCcnt-tracked, no VGPR staging) on
// CDNA5 toolchains that expose the builtin; plain copy otherwise.
__device__ __forceinline__ void stage16(unsigned short* lds_dst,
                                        const unsigned short* g_src) {
#if HAVE_ASYNC_LDS
    __builtin_amdgcn_global_load_async_to_lds_b128(
        (gintx4*)g_src, (lintx4*)lds_dst, 0, 0);
#else
    *(uint4*)lds_dst = *(const uint4*)g_src;
#endif
}

// Wait so that the *previous* tile's PER async copies per thread are complete
// (the PER just-issued ones for the next tile may remain outstanding), then
// workgroup-barrier so every wave's copies are visible.
template <int PER>
__device__ __forceinline__ void stage_wait(bool more_outstanding) {
#if HAVE_ASYNC_LDS
    if (more_outstanding) __builtin_amdgcn_s_wait_asynccnt(PER);
    else                  __builtin_amdgcn_s_wait_asynccnt(0);
#else
    (void)more_outstanding;
#endif
    __syncthreads();
}

// Load one 16x32 bf16 fragment from an LDS tile stored row-major [16][32].
__device__ __forceinline__ bf16x16 ld_frag(const unsigned short* base, int lane) {
    return *(const bf16x16*)(base + (lane & 15) * 32 + (lane >> 4) * 16);
}

__device__ __forceinline__ floatx8 wmma_bf16(bf16x16 a, bf16x16 b, floatx8 c) {
    return __builtin_amdgcn_wmma_f32_16x16x32_bf16(false, a, false, b,
                                                   (short)0, c, false, false);
}

// Monotonic grid barrier: target = nWG * (#barriers so far).  Counter memset
// to 0 by host each kernel_launch, so graph replays are deterministic.
__device__ __forceinline__ void gbar(unsigned int* cnt, unsigned int target) {
    __syncthreads();
    if (threadIdx.x == 0) {
        __threadfence();
        atomicAdd(cnt, 1u);
        while (__hip_atomic_load(cnt, __ATOMIC_ACQUIRE, __HIP_MEMORY_SCOPE_AGENT) < target) {
            __builtin_amdgcn_s_sleep(2);
        }
        __threadfence();
    }
    __syncthreads();
}

// ---------------------------------------------------------------------------
// Elementwise kernels
// ---------------------------------------------------------------------------
__global__ void cvt_bf16_kernel(const float* __restrict__ in,
                                unsigned short* __restrict__ out, long long n) {
    long long i = (long long)blockIdx.x * blockDim.x + threadIdx.x;
    if (i < n) out[i] = f2bf(in[i]);
}

__global__ void add_f32_kernel(const float* __restrict__ a, const float* __restrict__ b,
                               float* __restrict__ c, long long n) {
    long long i = (long long)blockIdx.x * blockDim.x + threadIdx.x;
    if (i < n) c[i] = a[i] + b[i];
}

__global__ void add_f32_bf_kernel(const float* __restrict__ a, const float* __restrict__ b,
                                  float* __restrict__ c, unsigned short* __restrict__ cb,
                                  long long n) {
    long long i = (long long)blockIdx.x * blockDim.x + threadIdx.x;
    if (i < n) { float v = a[i] + b[i]; c[i] = v; cb[i] = f2bf(v); }
}

// ---------------------------------------------------------------------------
// Generic WMMA GEMM:  C[M,N] = A[M,K] * W[N,K]^T (+ bias[n])
// A, W bf16 (as ushort).  Output f32 (Cf) and/or bf16 (Cb).
// Block tile 128(M) x 64(N), 256 threads = 8 waves, wave -> 16x64 strip.
// LDS double-buffered; next k-tile streamed (async) while current computes.
// Out-of-range rows/cols are clamped on load (never stored in epilogue).
// ---------------------------------------------------------------------------
__global__ void __launch_bounds__(256)
gemm_bf16_nt(const unsigned short* __restrict__ A,
             const unsigned short* __restrict__ Wt,
             const float* __restrict__ bias,
             float* __restrict__ Cf,
             unsigned short* __restrict__ Cb,
             int M, int N, int K) {
    __shared__ unsigned short As[2 * 128 * 32];
    __shared__ unsigned short Ws[2 * 64 * 32];
    const int tid  = threadIdx.x;
    const int lane = tid & 31;
    const int wave = tid >> 5;
    const int m0 = blockIdx.y * 128;
    const int n0 = blockIdx.x * 64;

    const int rowA = tid >> 1, segA = tid & 1;   // 128 rows x 2 halves of 16
    const int rowW = tid >> 2, segW = tid & 3;   // 64 rows x 4 segs of 8
    int gm = m0 + rowA; if (gm >= M) gm = M - 1;
    int gn = n0 + rowW; if (gn >= N) gn = N - 1;

    auto issue = [&](int k0, int buf) {
        const unsigned short* asrc = A + (size_t)gm * K + k0 + segA * 16;
        unsigned short* adst = As + buf * 4096 + rowA * 32 + segA * 16;
        stage16(adst,     asrc);
        stage16(adst + 8, asrc + 8);
        stage16(Ws + buf * 2048 + rowW * 32 + segW * 8,
                Wt + (size_t)gn * K + k0 + segW * 8);
    };

    floatx8 acc[4];
    for (int j = 0; j < 4; ++j) acc[j] = fzero8();

    const int KT = K / 32;
    issue(0, 0);
    for (int kt = 0; kt < KT; ++kt) {
        const int buf = kt & 1;
        if (kt + 1 < KT) issue((kt + 1) * 32, buf ^ 1);
        stage_wait<3>(kt + 1 < KT);

        const unsigned short* AsB = As + buf * 4096;
        const unsigned short* WsB = Ws + buf * 2048;
        bf16x16 a = ld_frag(AsB + wave * 512, lane);
        bf16x16 bfr[4];
#pragma unroll
        for (int nt = 0; nt < 4; ++nt) bfr[nt] = ld_frag(WsB + nt * 512, lane);
#pragma unroll
        for (int nt = 0; nt < 4; ++nt) acc[nt] = wmma_bf16(a, bfr[nt], acc[nt]);
        __syncthreads();
    }

    // epilogue: lanes 0-15 -> rows i, lanes 16-31 -> rows i+8
    const int ml = wave * 16 + (lane >> 4) * 8;
    const int nl = lane & 15;
#pragma unroll
    for (int nt = 0; nt < 4; ++nt) {
        int n = n0 + nt * 16 + nl;
        float bv = (bias && n < N) ? bias[n] : 0.f;
#pragma unroll
        for (int i = 0; i < 8; ++i) {
            int m = m0 + ml + i;
            if (m < M && n < N) {
                float v = acc[nt][i] + bv;
                if (Cf) Cf[(size_t)m * N + n] = v;
                if (Cb) Cb[(size_t)m * N + n] = f2bf(v);
            }
        }
    }
}

// ---------------------------------------------------------------------------
// One GRU timestep tile: this WG handles all 64 batch rows x 64 h-units at
// n0.  gh = h @ W_hh^T for chunks r/z/n via WMMA (double-buffered async
// staging), then fused gates.  gi_t: precomputed x-projection (bf16).
// ---------------------------------------------------------------------------
__device__ __forceinline__ void gru_step_body(
    const unsigned short* __restrict__ gi_t,    // [BATCH, 3H] bf16
    const unsigned short* __restrict__ hbf_in,  // [BATCH, HID] bf16
    const float* __restrict__ hf_in,            // [BATCH, HID] f32
    const unsigned short* __restrict__ Whh,     // [3H, HID] bf16
    const float* __restrict__ bhh,              // [3H]
    const float* __restrict__ mask_row,         // [BATCH] or null
    float* __restrict__ hf_out,
    unsigned short* __restrict__ hbf_out,
    float* __restrict__ out_row,                // [BATCH, HID] or null
    unsigned short* Hs /*[2*64*32]*/, unsigned short* Ws3 /*[2*3*64*32]*/,
    int n0) {

    const int tid  = threadIdx.x;
    const int lane = tid & 31;
    const int wave = tid >> 5;
    const int mt  = wave & 3;        // 4 m-tiles (64 batch rows)
    const int ntA = wave >> 2;       // wave handles n-tiles ntA and ntA+2
    const int row = tid >> 2, seg = tid & 3;

    auto issue = [&](int k0, int buf) {
        stage16(Hs + buf * 2048 + row * 32 + seg * 8,
                hbf_in + (size_t)row * HID + k0 + seg * 8);
#pragma unroll
        for (int c = 0; c < 3; ++c)
            stage16(Ws3 + buf * 6144 + c * 2048 + row * 32 + seg * 8,
                    Whh + (size_t)(c * HID + n0 + row) * HID + k0 + seg * 8);
    };

    floatx8 acc[3][2];
    for (int c = 0; c < 3; ++c)
        for (int j = 0; j < 2; ++j) acc[c][j] = fzero8();

    const int KT = HID / 32;
    issue(0, 0);
    for (int kt = 0; kt < KT; ++kt) {
        const int buf = kt & 1;
        if (kt + 1 < KT) issue((kt + 1) * 32, buf ^ 1);
        stage_wait<4>(kt + 1 < KT);

        const unsigned short* HsB = Hs + buf * 2048;
        const unsigned short* WsB = Ws3 + buf * 6144;
        bf16x16 a = ld_frag(HsB + mt * 512, lane);
        bf16x16 bfr[6];
#pragma unroll
        for (int c = 0; c < 3; ++c)
#pragma unroll
            for (int j = 0; j < 2; ++j)
                bfr[c * 2 + j] = ld_frag(WsB + c * 2048 + (ntA + 2 * j) * 512, lane);
#pragma unroll
        for (int u = 0; u < 6; ++u)
            acc[u >> 1][u & 1] = wmma_bf16(a, bfr[u], acc[u >> 1][u & 1]);
        __syncthreads();
    }

    const int mlb = mt * 16 + (lane >> 4) * 8;
    const int nl  = lane & 15;
#pragma unroll
    for (int j = 0; j < 2; ++j) {
        int nh = n0 + (ntA + 2 * j) * 16 + nl;
        float br = bhh[nh], bz = bhh[HID + nh], bn = bhh[2 * HID + nh];
#pragma unroll
        for (int i = 0; i < 8; ++i) {
            int b = mlb + i;
            size_t grow = (size_t)b * H3;
            float gr = bf2f(gi_t[grow + nh]);
            float gz = bf2f(gi_t[grow + HID + nh]);
            float gn = bf2f(gi_t[grow + 2 * HID + nh]);
            float r  = sigm(gr + acc[0][j][i] + br);
            float z  = sigm(gz + acc[1][j][i] + bz);
            float nv = tanhf(gn + r * (acc[2][j][i] + bn));
            float hold = hf_in[(size_t)b * HID + nh];
            float hn = (1.f - z) * nv + z * hold;
            if (mask_row) { float mv = mask_row[b]; hn = hn * mv + hold * (1.f - mv); }
            hf_out[(size_t)b * HID + nh]  = hn;
            hbf_out[(size_t)b * HID + nh] = f2bf(hn);
            if (out_row) out_row[(size_t)b * HID + nh] = hn;
        }
    }
}

// ---------------------------------------------------------------------------
// Persistent encoder leg: 16 WGs, loop over 512 timesteps, double-buffered h,
// one grid barrier per step.  Prefetches next step's gi block into L2.
// ---------------------------------------------------------------------------
__global__ void __launch_bounds__(256)
gru_encoder_persistent(const unsigned short* __restrict__ gi,
                       const unsigned short* __restrict__ Whh,
                       const float* __restrict__ bhh,
                       const float* __restrict__ mask,
                       float* __restrict__ h0f, float* __restrict__ h1f,
                       unsigned short* __restrict__ h0b, unsigned short* __restrict__ h1b,
                       float* __restrict__ out,
                       unsigned int* __restrict__ bar,
                       int reverse) {
    __shared__ unsigned short Hs[2 * 64 * 32];
    __shared__ unsigned short Ws3[2 * 3 * 64 * 32];
    const int n0 = blockIdx.x * 64;
    unsigned int gen = 0;
    for (int t = 0; t < SEQ; ++t) {
        int s = reverse ? (SEQ - 1 - t) : t;
        if (t + 1 < SEQ) {   // L2 prefetch of next step's gi (4096 thr x 96B)
            int sn = reverse ? (SEQ - 2 - t) : (t + 1);
            const unsigned short* g = gi + (size_t)sn * BATCH * H3
                + (size_t)(blockIdx.x * blockDim.x + threadIdx.x) * 48;
            __builtin_prefetch(g, 0, 1);
        }
        const float* hin          = (t & 1) ? h1f : h0f;
        float* hout               = (t & 1) ? h0f : h1f;
        const unsigned short* hbi = (t & 1) ? h1b : h0b;
        unsigned short* hbo       = (t & 1) ? h0b : h1b;
        gru_step_body(gi + (size_t)s * BATCH * H3, hbi, hin, Whh, bhh,
                      mask + (size_t)s * BATCH, hout, hbo,
                      out + (size_t)s * BATCH * HID, Hs, Ws3, n0);
        __threadfence();
        gbar(bar, gridDim.x * (++gen));
    }
}

// One-shot GRU step (write phase): hidden = gru_cell(m, hidden, Wr...)
__global__ void __launch_bounds__(256)
gru_step_once(const unsigned short* __restrict__ gi_t,
              const unsigned short* __restrict__ hbf_in,
              const float* __restrict__ hf_in,
              const unsigned short* __restrict__ Whh,
              const float* __restrict__ bhh,
              float* __restrict__ hf_out,
              unsigned short* __restrict__ hbf_out) {
    __shared__ unsigned short Hs[2 * 64 * 32];
    __shared__ unsigned short Ws3[2 * 3 * 64 * 32];
    gru_step_body(gi_t, hbf_in, hf_in, Whh, bhh, nullptr,
                  hf_out, hbf_out, nullptr, Hs, Ws3, blockIdx.x * 64);
}

// ---------------------------------------------------------------------------
// Write-phase attention over encoder outputs (one block per batch element):
// e[t] = <enc_sum[t,b,:], tvec[b,:]>, softmax over t, m = sum_t p_t*enc_sum.
// ---------------------------------------------------------------------------
__global__ void __launch_bounds__(256)
attn_write_kernel(const float* __restrict__ enc_sum,  // [SEQ,BATCH,HID]
                  const float* __restrict__ tvec,     // [BATCH,HID]
                  float* __restrict__ m_f,
                  unsigned short* __restrict__ m_bf,
                  float* __restrict__ memory,         // [BATCH,NSLOT,HID]
                  int slot) {
    __shared__ float es[SEQ];
    __shared__ float red[256];
    const int b = blockIdx.x, tid = threadIdx.x;

    for (int t = tid; t < SEQ; t += 256) {
        float a = 0.f;
        const float* er = enc_sum + ((size_t)t * BATCH + b) * HID;
        const float* tv = tvec + (size_t)b * HID;
        for (int k = 0; k < HID; ++k) a += er[k] * tv[k];
        es[t] = a;
    }
    __syncthreads();
    float mx = -3.4e38f;
    for (int t = tid; t < SEQ; t += 256) mx = fmaxf(mx, es[t]);
    red[tid] = mx; __syncthreads();
    for (int s = 128; s > 0; s >>= 1) { if (tid < s) red[tid] = fmaxf(red[tid], red[tid + s]); __syncthreads(); }
    mx = red[0]; __syncthreads();
    float sm = 0.f;
    for (int t = tid; t < SEQ; t += 256) { float e = __expf(es[t] - mx); es[t] = e; sm += e; }
    red[tid] = sm; __syncthreads();
    for (int s = 128; s > 0; s >>= 1) { if (tid < s) red[tid] += red[tid + s]; __syncthreads(); }
    float inv = 1.f / red[0]; __syncthreads();
    for (int t = tid; t < SEQ; t += 256) es[t] *= inv;
    __syncthreads();
    for (int k = tid; k < HID; k += 256) {
        float a = 0.f;
        for (int t = 0; t < SEQ; ++t) a += es[t] * enc_sum[((size_t)t * BATCH + b) * HID + k];
        m_f[(size_t)b * HID + k]  = a;
        m_bf[(size_t)b * HID + k] = f2bf(a);
        memory[((size_t)b * NSLOT + slot) * HID + k] = a;
    }
}

// ---------------------------------------------------------------------------
// Persistent decoder: GRU step -> proj (hn @ W_read^T) -> 3-slot attention
// -> out = read + hn.  3 grid barriers per timestep.
// ---------------------------------------------------------------------------
__global__ void __launch_bounds__(256)
gru_decoder_persistent(const unsigned short* __restrict__ gi,
                       const unsigned short* __restrict__ Whh,
                       const float* __restrict__ bhh,
                       const unsigned short* __restrict__ Wread,  // [HID,HID] bf16
                       const float* __restrict__ memf,            // [BATCH,NSLOT,HID]
                       float* __restrict__ h0f, float* __restrict__ h1f,
                       unsigned short* __restrict__ h0b, unsigned short* __restrict__ h1b,
                       float* __restrict__ tvec,                  // [BATCH,HID]
                       float* __restrict__ out,                   // [SEQ*BATCH,HID]
                       unsigned int* __restrict__ bar) {
    __shared__ unsigned short Hs[2 * 64 * 32];
    __shared__ unsigned short Ws3[2 * 3 * 64 * 32];
    __shared__ float red[256];
    __shared__ float pl[4][3];
    const int tid = threadIdx.x, lane = tid & 31, wave = tid >> 5;
    const int n0 = blockIdx.x * 64;
    const int row = tid >> 2, seg = tid & 3;
    unsigned int gen = 0;

    for (int t = 0; t < SEQ; ++t) {
        if (t + 1 < SEQ) {
            const unsigned short* g = gi + (size_t)(t + 1) * BATCH * H3
                + (size_t)(blockIdx.x * blockDim.x + threadIdx.x) * 48;
            __builtin_prefetch(g, 0, 1);
        }
        const float* hin          = (t & 1) ? h1f : h0f;
        float* hout               = (t & 1) ? h0f : h1f;
        const unsigned short* hbi = (t & 1) ? h1b : h0b;
        unsigned short* hbo       = (t & 1) ? h0b : h1b;
        gru_step_body(gi + (size_t)t * BATCH * H3, hbi, hin, Whh, bhh, nullptr,
                      hout, hbo, nullptr, Hs, Ws3, n0);
        __threadfence();
        gbar(bar, gridDim.x * (++gen));

        // --- projection: tvec = h_new @ W_read^T (WG's 64 columns) ---
        {
            const int mt = wave & 3, ntA = wave >> 2;
            floatx8 pacc[2];
            pacc[0] = fzero8(); pacc[1] = fzero8();

            auto pissue = [&](int k0, int buf) {
                stage16(Hs + buf * 2048 + row * 32 + seg * 8,
                        hbo + (size_t)row * HID + k0 + seg * 8);
                stage16(Ws3 + buf * 6144 + row * 32 + seg * 8,
                        Wread + (size_t)(n0 + row) * HID + k0 + seg * 8);
            };
            const int KT = HID / 32;
            pissue(0, 0);
            for (int kt = 0; kt < KT; ++kt) {
                const int buf = kt & 1;
                if (kt + 1 < KT) pissue((kt + 1) * 32, buf ^ 1);
                stage_wait<2>(kt + 1 < KT);
                bf16x16 a = ld_frag(Hs + buf * 2048 + mt * 512, lane);
                bf16x16 b0 = ld_frag(Ws3 + buf * 6144 + (ntA + 0) * 512, lane);
                bf16x16 b1 = ld_frag(Ws3 + buf * 6144 + (ntA + 2) * 512, lane);
                pacc[0] = wmma_bf16(a, b0, pacc[0]);
                pacc[1] = wmma_bf16(a, b1, pacc[1]);
                __syncthreads();
            }
            const int mlb = mt * 16 + (lane >> 4) * 8, nl = lane & 15;
#pragma unroll
            for (int j = 0; j < 2; ++j) {
                int nh = n0 + (ntA + 2 * j) * 16 + nl;
#pragma unroll
                for (int i = 0; i < 8; ++i)
                    tvec[(size_t)(mlb + i) * HID + nh] = pacc[j][i];
            }
        }
        __threadfence();
        gbar(bar, gridDim.x * (++gen));

        // --- attention over NSLOT memory slots for this WG's 4 batches ---
        for (int q = 0; q < 4; ++q) {
            int b = blockIdx.x * 4 + q;
            for (int j = 0; j < 3; ++j) {
                float p = 0.f;
                for (int k = tid; k < HID; k += 256)
                    p += memf[((size_t)b * NSLOT + j) * HID + k] * tvec[(size_t)b * HID + k];
                red[tid] = p; __syncthreads();
                for (int s = 128; s > 0; s >>= 1) { if (tid < s) red[tid] += red[tid + s]; __syncthreads(); }
                if (tid == 0) pl[q][j] = red[0];
                __syncthreads();
            }
        }
        if (tid < 4) {
            float e0 = pl[tid][0], e1 = pl[tid][1], e2 = pl[tid][2];
            float mx = fmaxf(e0, fmaxf(e1, e2));
            float x0 = __expf(e0 - mx), x1 = __expf(e1 - mx), x2 = __expf(e2 - mx);
            float inv = 1.f / (x0 + x1 + x2);
            pl[tid][0] = x0 * inv; pl[tid][1] = x1 * inv; pl[tid][2] = x2 * inv;
        }
        __syncthreads();
        for (int q = 0; q < 4; ++q) {
            int b = blockIdx.x * 4 + q;
            float p0 = pl[q][0], p1 = pl[q][1], p2 = pl[q][2];
            for (int k = tid; k < HID; k += 256) {
                float rv = p0 * memf[((size_t)b * NSLOT + 0) * HID + k]
                         + p1 * memf[((size_t)b * NSLOT + 1) * HID + k]
                         + p2 * memf[((size_t)b * NSLOT + 2) * HID + k];
                out[((size_t)t * BATCH + b) * HID + k] = rv + hout[(size_t)b * HID + k];
            }
        }
        __threadfence();
        gbar(bar, gridDim.x * (++gen));
    }
}

// ---------------------------------------------------------------------------
// Host orchestration
// ---------------------------------------------------------------------------
extern "C" void kernel_launch(void* const* d_in, const int* in_sizes, int n_in,
                              void* d_out, int out_size, void* d_ws, size_t ws_size,
                              hipStream_t stream) {
    (void)in_sizes; (void)n_in; (void)out_size; (void)ws_size;

    const float* emb     = (const float*)d_in[0];
    const float* mask    = (const float*)d_in[1];
    const float* Wf_ih   = (const float*)d_in[2];
    const float* Wf_hh   = (const float*)d_in[3];
    const float* bf_ih   = (const float*)d_in[4];
    const float* bf_hh   = (const float*)d_in[5];
    const float* Wb_ih   = (const float*)d_in[6];
    const float* Wb_hh   = (const float*)d_in[7];
    const float* bb_ih   = (const float*)d_in[8];
    const float* bb_hh   = (const float*)d_in[9];
    const float* Wr_ih   = (const float*)d_in[10];
    const float* Wr_hh   = (const float*)d_in[11];
    const float* br_ih   = (const float*)d_in[12];
    const float* br_hh   = (const float*)d_in[13];
    const float* Wd_ih   = (const float*)d_in[14];
    const float* Wd_hh   = (const float*)d_in[15];
    const float* bd_ih   = (const float*)d_in[16];
    const float* bd_hh   = (const float*)d_in[17];
    const float* W_write = (const float*)d_in[18];
    const float* W_read  = (const float*)d_in[19];
    float* out = (float*)d_out;

    const size_t SB = (size_t)SEQ * BATCH;

    // ---- workspace carve-out ----
    char* base = (char*)d_ws;
    size_t off = 0;
    auto alloc = [&](size_t bytes) -> void* {
        void* p = base + off;
        off += (bytes + 255) & ~(size_t)255;
        return p;
    };
    unsigned short* emb_bf = (unsigned short*)alloc(SB * EMB * 2);
    unsigned short* wfi = (unsigned short*)alloc((size_t)H3 * EMB * 2);
    unsigned short* wfh = (unsigned short*)alloc((size_t)H3 * HID * 2);
    unsigned short* wbi = (unsigned short*)alloc((size_t)H3 * EMB * 2);
    unsigned short* wbh = (unsigned short*)alloc((size_t)H3 * HID * 2);
    unsigned short* wri = (unsigned short*)alloc((size_t)H3 * EMB * 2);
    unsigned short* wrh = (unsigned short*)alloc((size_t)H3 * HID * 2);
    unsigned short* wdi = (unsigned short*)alloc((size_t)H3 * EMB * 2);
    unsigned short* wdh = (unsigned short*)alloc((size_t)H3 * HID * 2);
    unsigned short* wwr = (unsigned short*)alloc((size_t)HID * HID * 2);
    unsigned short* wrd = (unsigned short*)alloc((size_t)HID * HID * 2);
    unsigned short* gi_f = (unsigned short*)alloc(SB * H3 * 2);
    unsigned short* gi_b = (unsigned short*)alloc(SB * H3 * 2);
    unsigned short* gi_d = (unsigned short*)alloc(SB * H3 * 2);
    float* out_f   = (float*)alloc(SB * HID * 4);
    float* out_b   = (float*)alloc(SB * HID * 4);
    float* enc_sum = (float*)alloc(SB * HID * 4);

    const size_t HB  = (size_t)BATCH * HID;
    float* hf0 = (float*)alloc(HB * 4);  float* hf1 = (float*)alloc(HB * 4);
    float* hb0 = (float*)alloc(HB * 4);  float* hb1 = (float*)alloc(HB * 4);
    float* hd0 = (float*)alloc(HB * 4);  float* hd1 = (float*)alloc(HB * 4);
    float* hi0 = (float*)alloc(HB * 4);  float* hi1 = (float*)alloc(HB * 4);
    unsigned short* hf0b = (unsigned short*)alloc(HB * 2);
    unsigned short* hf1b = (unsigned short*)alloc(HB * 2);
    unsigned short* hb0b = (unsigned short*)alloc(HB * 2);
    unsigned short* hb1b = (unsigned short*)alloc(HB * 2);
    unsigned short* hd0b = (unsigned short*)alloc(HB * 2);
    unsigned short* hd1b = (unsigned short*)alloc(HB * 2);
    unsigned short* hi0b = (unsigned short*)alloc(HB * 2);
    unsigned short* hi1b = (unsigned short*)alloc(HB * 2);
    unsigned short* gi_w = (unsigned short*)alloc((size_t)BATCH * H3 * 2);
    float* tvec   = (float*)alloc(HB * 4);
    float* tvec_d = (float*)alloc(HB * 4);
    float* m_f = (float*)alloc(HB * 4);
    unsigned short* m_bf = (unsigned short*)alloc(HB * 2);
    float* memf = (float*)alloc((size_t)BATCH * NSLOT * HID * 4);
    unsigned int* bars = (unsigned int*)alloc(256);

    // ---- zero-init state (deterministic across graph replays) ----
    (void)hipMemsetAsync(hf0, 0, HB * 4, stream);  (void)hipMemsetAsync(hf1, 0, HB * 4, stream);
    (void)hipMemsetAsync(hb0, 0, HB * 4, stream);  (void)hipMemsetAsync(hb1, 0, HB * 4, stream);
    (void)hipMemsetAsync(hd0, 0, HB * 4, stream);  (void)hipMemsetAsync(hd1, 0, HB * 4, stream);
    (void)hipMemsetAsync(hf0b, 0, HB * 2, stream); (void)hipMemsetAsync(hf1b, 0, HB * 2, stream);
    (void)hipMemsetAsync(hb0b, 0, HB * 2, stream); (void)hipMemsetAsync(hb1b, 0, HB * 2, stream);
    (void)hipMemsetAsync(hd0b, 0, HB * 2, stream); (void)hipMemsetAsync(hd1b, 0, HB * 2, stream);
    (void)hipMemsetAsync(m_f, 0, HB * 4, stream);  (void)hipMemsetAsync(m_bf, 0, HB * 2, stream);
    (void)hipMemsetAsync(bars, 0, 256, stream);

    // ---- fp32 -> bf16 conversions ----
    auto cvt = [&](const float* s, unsigned short* d, long long n) {
        cvt_bf16_kernel<<<(unsigned)((n + 255) / 256), 256, 0, stream>>>(s, d, n);
    };
    cvt(emb, emb_bf, (long long)SB * EMB);
    cvt(Wf_ih, wfi, (long long)H3 * EMB); cvt(Wf_hh, wfh, (long long)H3 * HID);
    cvt(Wb_ih, wbi, (long long)H3 * EMB); cvt(Wb_hh, wbh, (long long)H3 * HID);
    cvt(Wr_ih, wri, (long long)H3 * EMB); cvt(Wr_hh, wrh, (long long)H3 * HID);
    cvt(Wd_ih, wdi, (long long)H3 * EMB); cvt(Wd_hh, wdh, (long long)H3 * HID);
    cvt(W_write, wwr, (long long)HID * HID); cvt(W_read, wrd, (long long)HID * HID);

    // ---- hoisted input projections: gi = emb @ W_ih^T + b_ih ----
    dim3 ggrid(H3 / 64, (unsigned)((SB + 127) / 128));
    gemm_bf16_nt<<<ggrid, 256, 0, stream>>>(emb_bf, wfi, bf_ih, nullptr, gi_f, (int)SB, H3, EMB);
    gemm_bf16_nt<<<ggrid, 256, 0, stream>>>(emb_bf, wbi, bb_ih, nullptr, gi_b, (int)SB, H3, EMB);
    gemm_bf16_nt<<<ggrid, 256, 0, stream>>>(emb_bf, wdi, bd_ih, nullptr, gi_d, (int)SB, H3, EMB);

    // ---- persistent encoder legs ----
    gru_encoder_persistent<<<HID / 64, 256, 0, stream>>>(
        gi_f, wfh, bf_hh, mask, hf0, hf1, hf0b, hf1b, out_f, bars + 0, 0);
    gru_encoder_persistent<<<HID / 64, 256, 0, stream>>>(
        gi_b, wbh, bb_hh, mask, hb0, hb1, hb0b, hb1b, out_b, bars + 16, 1);

    // enc_sum = out_f + out_b ; hidden = hf + hb  (final states live in buf 0)
    long long nEnc = (long long)SB * HID;
    add_f32_kernel<<<(unsigned)((nEnc + 255) / 256), 256, 0, stream>>>(out_f, out_b, enc_sum, nEnc);
    add_f32_bf_kernel<<<(unsigned)((HB + 255) / 256), 256, 0, stream>>>(hf0, hb0, hi0, hi0b, (long long)HB);

    // ---- write phase: build NSLOT memory slots ----
    float* hcF = hi0; unsigned short* hcB = hi0b;
    float* hnF = hi1; unsigned short* hnB = hi1b;
    for (int slot = 0; slot < NSLOT; ++slot) {
        dim3 g2(H3 / 64, 1);
        gemm_bf16_nt<<<g2, 256, 0, stream>>>(m_bf, wri, br_ih, nullptr, gi_w, BATCH, H3, EMB);
        gru_step_once<<<HID / 64, 256, 0, stream>>>(gi_w, hcB, hcF, wrh, br_hh, hnF, hnB);
        { float* tf = hcF; hcF = hnF; hnF = tf;
          unsigned short* tb = hcB; hcB = hnB; hnB = tb; }
        dim3 g3(HID / 64, 1);
        gemm_bf16_nt<<<g3, 256, 0, stream>>>(hcB, wwr, nullptr, tvec, nullptr, BATCH, HID, HID);
        attn_write_kernel<<<BATCH, 256, 0, stream>>>(enc_sum, tvec, m_f, m_bf, memf, slot);
    }

    // ---- persistent decoder (writes d_out directly) ----
    gru_decoder_persistent<<<HID / 64, 256, 0, stream>>>(
        gi_d, wdh, bd_hh, wrd, memf, hd0, hd1, hd0b, hd1b, tvec_d, out, bars + 32);
}
